// MDN_module_55791625175524
// MI455X (gfx1250) — matrix operand order repaired and checked by reading, not compiled, not run on hardware
//
#include <hip/hip_runtime.h>

typedef __attribute__((ext_vector_type(16))) _Float16 v16h;
typedef __attribute__((ext_vector_type(8)))  float    v8f;

#define BETA_C   0.99f
#define EPS_V_C  1e-3f
#define LOG2PI_C 1.8378770664093453f

// One wave handles 32 rows per iteration as two 16-row WMMA tiles.
// Layer1 (2->64, relu) on VALU directly into WMMA-A f16 layout,
// Layer2 (64->4) as two V_WMMA_F32_16X16X32_F16 accumulations.
__global__ __launch_bounds__(256)
void mdn_main(const float* __restrict__ x,
              const float* __restrict__ y,
              const float* __restrict__ eps,
              const float* __restrict__ W1,   // [2][64]
              const float* __restrict__ b1,   // [64]
              const float* __restrict__ W2,   // [64][4]
              const float* __restrict__ b2,   // [4]
              const float* __restrict__ Wv,   // [2][2]
              float* __restrict__ out_fx,     // [B*2]
              float* __restrict__ partials,   // [gridDim.x]
              int nrows)
{
    __shared__ float ct[8][32][4];   // per-wave C tile (32 rows x 4 cols)
    __shared__ float red[256];

    const int tid  = threadIdx.x;
    const int wave = tid >> 5;
    const int lane = tid & 31;
    const int m    = lane & 15;      // row within 16-row tile / B,C column
    const int g    = lane >> 4;      // lane-group bit (K sub-block select)
    const int n    = m;              // output column this lane owns in B/C

    // ---- per-lane layer-1 weights, ordered to match WMMA 16-bit A layout ----
    // A 16x32 f16: VGPR v<4 holds K = 2v+8g, 2v+1+8g ; v>=4 holds K = 2v+8+8g, +1
    float w0[32], w1c[32], wb[32];
#pragma unroll
    for (int c = 0; c < 2; ++c) {
#pragma unroll
        for (int v = 0; v < 8; ++v) {
#pragma unroll
            for (int d = 0; d < 2; ++d) {
                const int j = ((v < 4) ? 2 * v : 8 + 2 * v) + 8 * g + 32 * c + d;
                const int i = c * 16 + v * 2 + d;
                w0[i]  = W1[j];        // W1[0][j]
                w1c[i] = W1[64 + j];   // W1[1][j]
                wb[i]  = b1[j];
            }
        }
    }

    // ---- W2 as WMMA B matrices (32x16 f16 each): lane col n, K matches A layout (+32c) ----
    v16h bm0, bm1;
#pragma unroll
    for (int v = 0; v < 8; ++v) {
        const int k0 = ((v < 4) ? 2 * v : 8 + 2 * v) + 8 * g;
        const bool ok = (n < 4);
        const float a0 = ok ? W2[(k0     ) * 4 + n] : 0.0f;
        const float a1 = ok ? W2[(k0 +  1) * 4 + n] : 0.0f;
        const float a2 = ok ? W2[(k0 + 32) * 4 + n] : 0.0f;
        const float a3 = ok ? W2[(k0 + 33) * 4 + n] : 0.0f;
        bm0[2 * v]     = (_Float16)a0;
        bm0[2 * v + 1] = (_Float16)a1;
        bm1[2 * v]     = (_Float16)a2;
        bm1[2 * v + 1] = (_Float16)a3;
    }

    const float wv00 = Wv[0], wv01 = Wv[1], wv10 = Wv[2], wv11 = Wv[3];
    const float bc0 = b2[0], bc1 = b2[1], bc2 = b2[2], bc3 = b2[3];

    const float2* __restrict__ x2 = (const float2*)x;
    const float2* __restrict__ y2 = (const float2*)y;
    const float2* __restrict__ e2 = (const float2*)eps;

    float acc = 0.0f;

    const int nchunk  = nrows >> 5;          // 32 rows per chunk
    const int wstride = (int)(gridDim.x << 3);
    const int cstride = wstride << 5;        // row stride between a wave's chunks

    for (int chunk = (int)(blockIdx.x * 8 + wave); chunk < nchunk; chunk += wstride) {
        const int base = chunk << 5;
        const int erow = base + lane;        // epilogue row owned by this lane

        // ---- issue ALL per-chunk global loads up front (hide HBM latency) ----
        const float2 xv0 = x2[base + m];
        const float2 xv1 = x2[base + 16 + m];
        const float2 ev  = e2[erow];
        const float2 yv  = y2[erow];

        // ---- prefetch next chunk's input streams (global_prefetch_b8) ----
        const int nb = base + cstride;
        if (nb < nrows) {
            __builtin_prefetch((const char*)(x2 + nb) + lane * 8, 0, 3);
            __builtin_prefetch((const char*)(e2 + nb) + lane * 8, 0, 3);
            __builtin_prefetch((const char*)(y2 + nb) + lane * 8, 0, 3);
        }

#pragma unroll
        for (int t = 0; t < 2; ++t) {
            const float xlo = t ? xv1.x : xv0.x;
            const float xhi = t ? xv1.y : xv0.y;

            v8f cc = {0.f, 0.f, 0.f, 0.f, 0.f, 0.f, 0.f, 0.f};

            v16h a;
#pragma unroll
            for (int v = 0; v < 8; ++v) {
                float h0 = fmaf(xhi, w1c[2 * v],     fmaf(xlo, w0[2 * v],     wb[2 * v]));
                float h1 = fmaf(xhi, w1c[2 * v + 1], fmaf(xlo, w0[2 * v + 1], wb[2 * v + 1]));
                h0 = fmaxf(h0, 0.0f);
                h1 = fmaxf(h1, 0.0f);
                a[2 * v]     = (_Float16)h0;
                a[2 * v + 1] = (_Float16)h1;
            }
            cc = __builtin_amdgcn_wmma_f32_16x16x32_f16(false, a, false, bm0,
                                                        (short)0, cc, false, false);
#pragma unroll
            for (int v = 0; v < 8; ++v) {
                const int i = 16 + 2 * v;
                float h0 = fmaf(xhi, w1c[i],     fmaf(xlo, w0[i],     wb[i]));
                float h1 = fmaf(xhi, w1c[i + 1], fmaf(xlo, w0[i + 1], wb[i + 1]));
                h0 = fmaxf(h0, 0.0f);
                h1 = fmaxf(h1, 0.0f);
                a[2 * v]     = (_Float16)h0;
                a[2 * v + 1] = (_Float16)h1;
            }
            cc = __builtin_amdgcn_wmma_f32_16x16x32_f16(false, a, false, bm1,
                                                        (short)0, cc, false, false);

            // C layout: lane n (both halves), VGPR r -> row r + 8g
            if (n < 4) {
#pragma unroll
                for (int r = 0; r < 8; ++r)
                    ct[wave][t * 16 + r + 8 * g][n] = cc[r];
            }
        }
        __syncthreads();

        // ---- epilogue: one row per lane ----
        {
            const float o0  = ct[wave][lane][0] + bc0;   // mu head
            const float o1  = ct[wave][lane][1] + bc1;
            const float lv0 = ct[wave][lane][2] + bc2;   // log-var head
            const float lv1 = ct[wave][lane][3] + bc3;

            const float xa = g ? xv1.x : xv0.x;
            const float xb = g ? xv1.y : xv0.y;

            // Vx = ||x Wv||^2 + eps ; Vmu likewise
            const float zx0 = fmaf(xb, wv10, xa * wv00);
            const float zx1 = fmaf(xb, wv11, xa * wv01);
            const float Vx  = fmaf(zx0, zx0, zx1 * zx1) + EPS_V_C;
            const float zm0 = fmaf(o1, wv10, o0 * wv00);
            const float zm1 = fmaf(o1, wv11, o0 * wv01);
            const float Vmu = fmaf(zm0, zm0, zm1 * zm1) + EPS_V_C;

            // beta*Vx - relu(beta*Vx - Vmu) == min(beta*Vx, Vmu)
            const float scale = fminf(BETA_C * Vx, Vmu) * __builtin_amdgcn_rcpf(Vmu);
            const float mu0 = o0 * scale;
            const float mu1 = o1 * scale;

            const float sd0 = __expf(0.5f * lv0);   // sqrt(var)
            const float sd1 = __expf(0.5f * lv1);
            const float iv0 = __expf(-lv0);         // 1/var
            const float iv1 = __expf(-lv1);

            float2 fx;
            fx.x = fmaf(sd0, ev.x, mu0);
            fx.y = fmaf(sd1, ev.y, mu1);
            ((float2*)out_fx)[erow] = fx;

            const float d0 = yv.x - mu0;
            const float d1 = yv.y - mu1;
            const float term = fmaf(d0 * d0, iv0,
                              fmaf(d1 * d1, iv1, lv0 + lv1 + 2.0f * LOG2PI_C));
            acc = fmaf(0.5f, term, acc);   // contributes -logp
        }
        __syncthreads();
    }

    // deterministic block reduction
    red[tid] = acc;
    __syncthreads();
#pragma unroll
    for (int off = 128; off > 0; off >>= 1) {
        if (tid < off) red[tid] += red[tid + off];
        __syncthreads();
    }
    if (tid == 0) partials[blockIdx.x] = red[0];
}

__global__ __launch_bounds__(256)
void mdn_reduce(const float* __restrict__ partials, int nparts,
                float* __restrict__ out_scalar)
{
    __shared__ float red[256];
    float s = 0.0f;
    for (int i = (int)threadIdx.x; i < nparts; i += 256) s += partials[i];
    red[threadIdx.x] = s;
    __syncthreads();
#pragma unroll
    for (int off = 128; off > 0; off >>= 1) {
        if ((int)threadIdx.x < off) red[threadIdx.x] += red[threadIdx.x + off];
        __syncthreads();
    }
    if (threadIdx.x == 0) *out_scalar = red[0];
}

extern "C" void kernel_launch(void* const* d_in, const int* in_sizes, int n_in,
                              void* d_out, int out_size, void* d_ws, size_t ws_size,
                              hipStream_t stream)
{
    const float* x   = (const float*)d_in[0];
    const float* y   = (const float*)d_in[1];
    const float* eps = (const float*)d_in[2];
    const float* W1  = (const float*)d_in[3];
    const float* b1  = (const float*)d_in[4];
    const float* W2  = (const float*)d_in[5];
    const float* b2  = (const float*)d_in[6];
    const float* Wv  = (const float*)d_in[7];

    const int B = in_sizes[0] / 2;          // x is [B,2]
    float* out      = (float*)d_out;
    float* partials = (float*)d_ws;

    const int grid = 1024;                  // 8192 waves; 65536 chunks -> 8 uniform iters
    mdn_main<<<grid, 256, 0, stream>>>(x, y, eps, W1, b1, W2, b2, Wv,
                                       out, partials, B);
    mdn_reduce<<<1, 256, 0, stream>>>(partials, grid, out + (size_t)2 * B);
}
